// MoE_2508260901294
// MI455X (gfx1250) — matrix operand order repaired and checked by reading, not compiled, not run on hardware
//
#include <hip/hip_runtime.h>
#include <cstdint>
#include <cstddef>

// ---------------- Config (matches reference) ----------------
#define D_MODEL 1024
#define D_MLP   4096
#define N_EXP   8
#define TOKENS  2048

// GEMM tiling
#define BM 128          // rows per block tile
#define BN 64           // cols per block tile
#define KT 32           // K per WMMA step
#define LDA 40          // LDS row stride for A tile (bf16 elems, 16B-aligned rows)
#define LDB 80          // LDS row stride for B tile (bf16 elems, 16B-aligned rows)

typedef unsigned short u16;
typedef __attribute__((ext_vector_type(16))) __bf16 v16bf;
typedef __attribute__((ext_vector_type(8)))  __bf16 v8bf;
typedef __attribute__((ext_vector_type(2)))  __bf16 v2bf;
typedef __attribute__((ext_vector_type(8)))  float  v8f;
typedef __attribute__((ext_vector_type(2)))  float  v2f;
typedef __attribute__((ext_vector_type(4)))  int    v4i;

union FragAB { v16bf v; uint4 q[2]; };

__device__ __forceinline__ int mini(int a, int b) { return a < b ? a : b; }

// ---- fp32 -> bf16 via vector fptrunc (RNE); backend picks native packed cvt ----
__device__ __forceinline__ unsigned pk2(float a, float b) {
  v2f f; f.x = a; f.y = b;
  v2bf r = __builtin_convertvector(f, v2bf);
  return __builtin_bit_cast(unsigned, r);
}
__device__ __forceinline__ uint4 cvt8(const float4& a, const float4& b) {
  v8f f;
  f[0] = a.x; f[1] = a.y; f[2] = a.z; f[3] = a.w;
  f[4] = b.x; f[5] = b.y; f[6] = b.z; f[7] = b.w;
  v8bf r = __builtin_convertvector(f, v8bf);
  return __builtin_bit_cast(uint4, r);
}

// ---- CDNA5 async global->LDS copy (ASYNCcnt path) ----
// ROCm 7.2 signature (probe-verified): (v4i addrspace(1)*, v4i addrspace(3)*, imm, imm)
#if __has_builtin(__builtin_amdgcn_global_load_async_to_lds_b128)
#define HAVE_ASYNC_LDS 1
#else
#define HAVE_ASYNC_LDS 0
#endif

#if HAVE_ASYNC_LDS
__device__ __forceinline__ void async_cp16(const u16* g, u16* l) {
  __builtin_amdgcn_global_load_async_to_lds_b128(
      (__attribute__((address_space(1))) v4i*)g,
      (__attribute__((address_space(3))) v4i*)l,
      0, 0);
}
__device__ __forceinline__ void async_wait0() {
#if __has_builtin(__builtin_amdgcn_s_wait_asynccnt)
  __builtin_amdgcn_s_wait_asynccnt(0);
#else
  asm volatile("s_wait_asynccnt 0" ::: "memory");
#endif
}
#endif

// ---------------- 1) Router: logits -> softmax -> top-2 -> renorm ----------------
// one wave32 per token
__global__ void moe_router(const float* __restrict__ x, const float* __restrict__ Wr,
                           int2* __restrict__ top_e, float2* __restrict__ top_w) {
  const int wave = threadIdx.x >> 5;
  const int lane = threadIdx.x & 31;
  const int t = blockIdx.x * 8 + wave;
  if (t >= TOKENS) return;
  const float* hr = x + (size_t)t * D_MODEL;

  float acc[N_EXP] = {0.f, 0.f, 0.f, 0.f, 0.f, 0.f, 0.f, 0.f};
  for (int d = lane; d < D_MODEL; d += 32) {
    float hv = hr[d];
    const float4* wr = (const float4*)(Wr + (size_t)d * N_EXP);
    float4 w0 = wr[0], w1 = wr[1];
    acc[0] += hv * w0.x; acc[1] += hv * w0.y; acc[2] += hv * w0.z; acc[3] += hv * w0.w;
    acc[4] += hv * w1.x; acc[5] += hv * w1.y; acc[6] += hv * w1.z; acc[7] += hv * w1.w;
  }
#pragma unroll
  for (int e = 0; e < N_EXP; ++e) {
#pragma unroll
    for (int off = 16; off; off >>= 1) acc[e] += __shfl_xor(acc[e], off, 32);
  }
  if (lane == 0) {
    float m = acc[0];
#pragma unroll
    for (int e = 1; e < N_EXP; ++e) m = fmaxf(m, acc[e]);
    float p[N_EXP];
#pragma unroll
    for (int e = 0; e < N_EXP; ++e) p[e] = __expf(acc[e] - m);
    int e0 = 0; float p0 = p[0];
#pragma unroll
    for (int e = 1; e < N_EXP; ++e) if (p[e] > p0) { p0 = p[e]; e0 = e; }
    int e1 = (e0 == 0) ? 1 : 0; float p1 = p[e1];
#pragma unroll
    for (int e = 0; e < N_EXP; ++e)
      if (e != e0 && p[e] > p1) { p1 = p[e]; e1 = e; }
    float s = 1.0f / (p0 + p1);
    int2 te; te.x = e0; te.y = e1;
    float2 tw; tw.x = p0 * s; tw.y = p1 * s;
    top_e[t] = te; top_w[t] = tw;
  }
}

// ---------------- 2) Deterministic per-expert token lists ----------------
__global__ void moe_build(const int2* __restrict__ te, const float2* __restrict__ tw,
                          int* __restrict__ idx, float* __restrict__ gw,
                          int* __restrict__ counts) {
  const int e = threadIdx.x;
  if (e >= N_EXP) return;
  int c = 0;
  for (int t = 0; t < TOKENS; ++t) {
    int2 a = te[t];
    float2 w = tw[t];
    if (a.x == e)      { idx[e * TOKENS + c] = t; gw[e * TOKENS + c] = w.x; ++c; }
    else if (a.y == e) { idx[e * TOKENS + c] = t; gw[e * TOKENS + c] = w.y; ++c; }
  }
  counts[e] = c;
}

// ---------------- 3) zero output + convert activations to bf16 ----------------
__global__ void moe_prep(const float* __restrict__ x, u16* __restrict__ hbf,
                         float* __restrict__ out) {
  const int n4 = TOKENS * D_MODEL / 4;
  for (int i = blockIdx.x * blockDim.x + threadIdx.x; i < n4;
       i += gridDim.x * blockDim.x) {
    float4 v = ((const float4*)x)[i];
    uint2 b; b.x = pk2(v.x, v.y); b.y = pk2(v.z, v.w);
    ((uint2*)hbf)[i] = b;
    float4 z; z.x = z.y = z.z = z.w = 0.f;
    ((float4*)out)[i] = z;
  }
}

// ---------------- 4) GEMM1: hid = silu(h@Wg) * (h@Wi), gathered rows ----------------
// block = 256 threads (8 waves). Block tile 128x64, wave tile 32x32 (2x2 wmma frags).
__global__ __launch_bounds__(256) void moe_gemm1(
    const u16* __restrict__ hbf, const float* __restrict__ Wg,
    const float* __restrict__ Wi, const int* __restrict__ idx,
    const int* __restrict__ cntp, u16* __restrict__ hid) {
  const int cnt = *cntp;
  const int m0 = blockIdx.y * BM;
  if (m0 >= cnt) return;
  const int n0 = blockIdx.x * BN;

  __shared__ u16 lA[2][BM * LDA];
  __shared__ u16 lG[2][KT * LDB];
  __shared__ u16 lI[2][KT * LDB];

  const int tid  = threadIdx.x;
  const int lane = tid & 31;
  const int wave = tid >> 5;
  const int wm = (wave & 3) * 32;   // wave M offset in tile
  const int wn = (wave >> 2) * 32;  // wave N offset in tile

  // A-tile loader mapping: 2 threads per row, 16 bf16 each
  const int arow = tid >> 1;
  const int aseg = (tid & 1) * 16;
  const int tok  = idx[mini(m0 + arow, cnt - 1)];
  const u16* aptr = hbf + (size_t)tok * D_MODEL + aseg;

  // weight-tile loader mapping: 8 consecutive fp32 per thread
  const int wrow = (tid * 8) >> 6;  // 0..31  (K row in tile)
  const int wcol = (tid * 8) & 63;  // 0..56
  const float* gp0 = Wg + (size_t)wrow * D_MLP + n0 + wcol;
  const float* ip0 = Wi + (size_t)wrow * D_MLP + n0 + wcol;

  { // stage k-tile 0 -> buffer 0
#if HAVE_ASYNC_LDS
    async_cp16(aptr,     &lA[0][arow * LDA + aseg]);
    async_cp16(aptr + 8, &lA[0][arow * LDA + aseg + 8]);
#else
    uint4 a0 = *(const uint4*)(aptr);
    uint4 a1 = *(const uint4*)(aptr + 8);
    *(uint4*)&lA[0][arow * LDA + aseg]     = a0;
    *(uint4*)&lA[0][arow * LDA + aseg + 8] = a1;
#endif
    float4 g0 = *(const float4*)(gp0), g1 = *(const float4*)(gp0 + 4);
    float4 i0 = *(const float4*)(ip0), i1 = *(const float4*)(ip0 + 4);
    *(uint4*)&lG[0][wrow * LDB + wcol] = cvt8(g0, g1);
    *(uint4*)&lI[0][wrow * LDB + wcol] = cvt8(i0, i1);
#if HAVE_ASYNC_LDS
    async_wait0();
#endif
  }
  __syncthreads();

  const v8f vzero = {0.f, 0.f, 0.f, 0.f, 0.f, 0.f, 0.f, 0.f};
  v8f accG[2][2], accI[2][2];
#pragma unroll
  for (int a = 0; a < 2; ++a)
#pragma unroll
    for (int b = 0; b < 2; ++b) { accG[a][b] = vzero; accI[a][b] = vzero; }

  int buf = 0;
  const int KSTEPS = D_MODEL / KT;  // 32
  for (int kt = 0; kt < KSTEPS; ++kt) {
#if !HAVE_ASYNC_LDS
    uint4 ra0, ra1;
#endif
    uint4 rg, ri;
    const bool more = (kt + 1) < KSTEPS;
    if (more) {  // stage next tile while WMMAs run
      const u16* ap = aptr + (kt + 1) * KT;
#if HAVE_ASYNC_LDS
      async_cp16(ap,     &lA[buf ^ 1][arow * LDA + aseg]);
      async_cp16(ap + 8, &lA[buf ^ 1][arow * LDA + aseg + 8]);
#else
      ra0 = *(const uint4*)(ap);
      ra1 = *(const uint4*)(ap + 8);
#endif
      const float* gp = gp0 + (size_t)(kt + 1) * KT * D_MLP;
      const float* ip = ip0 + (size_t)(kt + 1) * KT * D_MLP;
      float4 g0 = *(const float4*)gp, g1 = *(const float4*)(gp + 4);
      float4 i0 = *(const float4*)ip, i1 = *(const float4*)(ip + 4);
      rg = cvt8(g0, g1); ri = cvt8(i0, i1);
      if (kt + 3 < KSTEPS) {  // warm L2 two tiles ahead (global_prefetch_b8)
        __builtin_prefetch(gp0 + (size_t)(kt + 3) * KT * D_MLP, 0, 0);
        __builtin_prefetch(ip0 + (size_t)(kt + 3) * KT * D_MLP, 0, 0);
      }
    }

    // A frags per ISA 7.12.2: lane<16 -> K {0..7,16..23}, lane>=16 -> {8..15,24..31}
    FragAB fa[2];
    const int aK = (lane >> 4) * 8;
#pragma unroll
    for (int ms = 0; ms < 2; ++ms) {
      const int aM = wm + ms * 16 + (lane & 15);
      fa[ms].q[0] = *(const uint4*)&lA[buf][aM * LDA + aK];
      fa[ms].q[1] = *(const uint4*)&lA[buf][aM * LDA + aK + 16];
    }
#pragma unroll
    for (int ns = 0; ns < 2; ++ns) {
      const int bN = wn + ns * 16;   // B frag: lane = N col, 16 K values in A's K order
      FragAB fg, fi;
#pragma unroll
      for (int s = 0; s < 16; ++s) {
        const int bk = (lane >> 4) * 8 + (s & 7) + (s >> 3) * 16;
        fg.v[s] = __builtin_bit_cast(__bf16, lG[buf][bk * LDB + bN + (lane & 15)]);
        fi.v[s] = __builtin_bit_cast(__bf16, lI[buf][bk * LDB + bN + (lane & 15)]);
      }
#pragma unroll
      for (int ms = 0; ms < 2; ++ms) {
        accG[ms][ns] = __builtin_amdgcn_wmma_f32_16x16x32_bf16(
            false, fa[ms].v, false, fg.v, (short)0, accG[ms][ns], false, false);
        accI[ms][ns] = __builtin_amdgcn_wmma_f32_16x16x32_bf16(
            false, fa[ms].v, false, fi.v, (short)0, accI[ms][ns], false, false);
      }
    }

    if (more) {
      buf ^= 1;
#if !HAVE_ASYNC_LDS
      *(uint4*)&lA[buf][arow * LDA + aseg]     = ra0;
      *(uint4*)&lA[buf][arow * LDA + aseg + 8] = ra1;
#endif
      *(uint4*)&lG[buf][wrow * LDB + wcol] = rg;
      *(uint4*)&lI[buf][wrow * LDB + wcol] = ri;
#if HAVE_ASYNC_LDS
      async_wait0();
#endif
    }
    __syncthreads();
  }

  // epilogue: silu(gate) * in -> bf16 hid (compacted rows)
  const int mhalf = (lane >> 4) * 8;  // C layout: VGPR i -> M=i (lo lanes) / M=8+i (hi lanes)
#pragma unroll
  for (int ms = 0; ms < 2; ++ms) {
#pragma unroll
    for (int ns = 0; ns < 2; ++ns) {
      const int col = n0 + wn + ns * 16 + (lane & 15);
#pragma unroll
      for (int i = 0; i < 8; i += 2) {
        const int m = m0 + wm + ms * 16 + mhalf + i;
        float g0 = accG[ms][ns][i],     v0 = accI[ms][ns][i];
        float g1 = accG[ms][ns][i + 1], v1 = accI[ms][ns][i + 1];
        float s0 = (g0 / (1.0f + __expf(-g0))) * v0;
        float s1 = (g1 / (1.0f + __expf(-g1))) * v1;
        unsigned pk = pk2(s0, s1);
        if (m < cnt)     hid[(size_t)m * D_MLP + col]       = (u16)(pk & 0xFFFFu);
        if (m + 1 < cnt) hid[(size_t)(m + 1) * D_MLP + col] = (u16)(pk >> 16);
      }
    }
  }
}

// ---------------- 5) GEMM2: out[token] += (hid @ Wo) * gate_weight ----------------
__global__ __launch_bounds__(256) void moe_gemm2(
    const u16* __restrict__ hid, const float* __restrict__ Wo,
    const int* __restrict__ idx, const float* __restrict__ gw,
    const int* __restrict__ cntp, float* __restrict__ out) {
  const int cnt = *cntp;
  const int m0 = blockIdx.y * BM;
  if (m0 >= cnt) return;
  const int n0 = blockIdx.x * BN;

  __shared__ u16 lA[2][BM * LDA];
  __shared__ u16 lB[2][KT * LDB];

  const int tid  = threadIdx.x;
  const int lane = tid & 31;
  const int wave = tid >> 5;
  const int wm = (wave & 3) * 32;
  const int wn = (wave >> 2) * 32;

  const int arow = tid >> 1;
  const int aseg = (tid & 1) * 16;
  const int amr  = mini(m0 + arow, cnt - 1);
  const u16* aptr = hid + (size_t)amr * D_MLP + aseg;

  const int wrow = (tid * 8) >> 6;
  const int wcol = (tid * 8) & 63;
  const float* bp0 = Wo + (size_t)wrow * D_MODEL + n0 + wcol;

  {
#if HAVE_ASYNC_LDS
    async_cp16(aptr,     &lA[0][arow * LDA + aseg]);
    async_cp16(aptr + 8, &lA[0][arow * LDA + aseg + 8]);
#else
    uint4 a0 = *(const uint4*)(aptr);
    uint4 a1 = *(const uint4*)(aptr + 8);
    *(uint4*)&lA[0][arow * LDA + aseg]     = a0;
    *(uint4*)&lA[0][arow * LDA + aseg + 8] = a1;
#endif
    float4 b0 = *(const float4*)(bp0), b1 = *(const float4*)(bp0 + 4);
    *(uint4*)&lB[0][wrow * LDB + wcol] = cvt8(b0, b1);
#if HAVE_ASYNC_LDS
    async_wait0();
#endif
  }
  __syncthreads();

  const v8f vzero = {0.f, 0.f, 0.f, 0.f, 0.f, 0.f, 0.f, 0.f};
  v8f acc[2][2];
#pragma unroll
  for (int a = 0; a < 2; ++a)
#pragma unroll
    for (int b = 0; b < 2; ++b) acc[a][b] = vzero;

  int buf = 0;
  const int KSTEPS = D_MLP / KT;  // 128
  for (int kt = 0; kt < KSTEPS; ++kt) {
#if !HAVE_ASYNC_LDS
    uint4 ra0, ra1;
#endif
    uint4 rb;
    const bool more = (kt + 1) < KSTEPS;
    if (more) {
      const u16* ap = aptr + (kt + 1) * KT;
#if HAVE_ASYNC_LDS
      async_cp16(ap,     &lA[buf ^ 1][arow * LDA + aseg]);
      async_cp16(ap + 8, &lA[buf ^ 1][arow * LDA + aseg + 8]);
#else
      ra0 = *(const uint4*)(ap);
      ra1 = *(const uint4*)(ap + 8);
#endif
      const float* bp = bp0 + (size_t)(kt + 1) * KT * D_MODEL;
      float4 b0 = *(const float4*)bp, b1 = *(const float4*)(bp + 4);
      rb = cvt8(b0, b1);
      if (kt + 3 < KSTEPS)
        __builtin_prefetch(bp0 + (size_t)(kt + 3) * KT * D_MODEL, 0, 0);
    }

    FragAB fa[2];
    const int aK = (lane >> 4) * 8;
#pragma unroll
    for (int ms = 0; ms < 2; ++ms) {
      const int aM = wm + ms * 16 + (lane & 15);
      fa[ms].q[0] = *(const uint4*)&lA[buf][aM * LDA + aK];
      fa[ms].q[1] = *(const uint4*)&lA[buf][aM * LDA + aK + 16];
    }
#pragma unroll
    for (int ns = 0; ns < 2; ++ns) {
      const int bN = wn + ns * 16;
      FragAB fb;
#pragma unroll
      for (int s = 0; s < 16; ++s) {
        const int bk = (lane >> 4) * 8 + (s & 7) + (s >> 3) * 16;
        fb.v[s] = __builtin_bit_cast(__bf16, lB[buf][bk * LDB + bN + (lane & 15)]);
      }
#pragma unroll
      for (int ms = 0; ms < 2; ++ms) {
        acc[ms][ns] = __builtin_amdgcn_wmma_f32_16x16x32_bf16(
            false, fa[ms].v, false, fb.v, (short)0, acc[ms][ns], false, false);
      }
    }

    if (more) {
      buf ^= 1;
#if !HAVE_ASYNC_LDS
      *(uint4*)&lA[buf][arow * LDA + aseg]     = ra0;
      *(uint4*)&lA[buf][arow * LDA + aseg + 8] = ra1;
#endif
      *(uint4*)&lB[buf][wrow * LDB + wcol] = rb;
#if HAVE_ASYNC_LDS
      async_wait0();
#endif
    }
    __syncthreads();
  }

  // epilogue: scale by routing weight, scatter-accumulate into out[token]
  const int mhalf = (lane >> 4) * 8;
#pragma unroll
  for (int ms = 0; ms < 2; ++ms) {
#pragma unroll
    for (int ns = 0; ns < 2; ++ns) {
      const int col = n0 + wn + ns * 16 + (lane & 15);
#pragma unroll
      for (int i = 0; i < 8; ++i) {
        const int m = m0 + wm + ms * 16 + mhalf + i;
        if (m < cnt) {
          const int tokr = idx[m];
          const float w  = gw[m];
          float* o = out + (size_t)tokr * D_MODEL + col;
          *o += acc[ms][ns][i] * w;  // unique rows per expert; launches stream-serialized
        }
      }
    }
  }
}

// ---------------- host-side launch ----------------
extern "C" void kernel_launch(void* const* d_in, const int* in_sizes, int n_in,
                              void* d_out, int out_size, void* d_ws, size_t ws_size,
                              hipStream_t stream) {
  const float* x   = (const float*)d_in[0];
  const float* Wr  = (const float*)d_in[1];
  const float* Weg = (const float*)d_in[2];
  const float* Wei = (const float*)d_in[3];
  const float* Weo = (const float*)d_in[4];
  float* out = (float*)d_out;

  char* wsb = (char*)d_ws;
  size_t off = 0;
  auto walloc = [&](size_t bytes) -> void* {
    void* p = wsb + off;
    off += (bytes + 255) & ~(size_t)255;
    return p;
  };
  int*    counts = (int*)walloc(N_EXP * sizeof(int));
  int2*   top_e  = (int2*)walloc((size_t)TOKENS * sizeof(int2));
  float2* top_w  = (float2*)walloc((size_t)TOKENS * sizeof(float2));
  int*    idx    = (int*)walloc((size_t)N_EXP * TOKENS * sizeof(int));
  float*  gw     = (float*)walloc((size_t)N_EXP * TOKENS * sizeof(float));
  u16*    hbf    = (u16*)walloc((size_t)TOKENS * D_MODEL * sizeof(u16));
  u16*    hid    = (u16*)walloc((size_t)TOKENS * D_MLP * sizeof(u16));
  (void)ws_size; (void)in_sizes; (void)n_in; (void)out_size;
  // total workspace ~20.3 MB

  moe_router<<<TOKENS / 8, 256, 0, stream>>>(x, Wr, top_e, top_w);
  moe_build<<<1, 32, 0, stream>>>(top_e, top_w, idx, gw, counts);
  moe_prep<<<1024, 256, 0, stream>>>(x, hbf, out);

  for (int e = 0; e < N_EXP; ++e) {
    const float* Wg = Weg + (size_t)e * D_MODEL * D_MLP;
    const float* Wi = Wei + (size_t)e * D_MODEL * D_MLP;
    const float* Wo = Weo + (size_t)e * D_MLP * D_MODEL;
    moe_gemm1<<<dim3(D_MLP / BN, TOKENS / BM), 256, 0, stream>>>(
        hbf, Wg, Wi, idx + (size_t)e * TOKENS, counts + e, hid);
    moe_gemm2<<<dim3(D_MODEL / BN, TOKENS / BM), 256, 0, stream>>>(
        hid, Wo, idx + (size_t)e * TOKENS, gw + (size_t)e * TOKENS, counts + e, out);
  }
}
